// CABlock_85942295593250
// MI455X (gfx1250) — matrix operand order repaired and checked by reading, not compiled, not run on hardware
//
#include <hip/hip_runtime.h>
#include <hip/hip_bf16.h>
#include <math.h>

// ---------------- problem constants ----------------
#define Bn   64
#define Nn   784
#define Dn   768
#define Hn   32
#define HDn  24
#define Gn   2
#define HIDn 3072
#define Mn   785              // N+1
#define MTn  (Bn*Mn)          // 50240 rows of the concat
#define SCALEq 0.20412414523193154f   // 24^-0.5

// ---------------- workspace layout (bytes, all 256B aligned) ----------------
static const size_t SZ_MTD_H = (size_t)MTn * Dn * 2;     // 77,168,640
static const size_t OFF_NH  = 0;
static const size_t OFF_KH  = OFF_NH + SZ_MTD_H;
static const size_t OFF_VH  = OFF_KH + SZ_MTD_H;
static const size_t OFF_QH  = OFF_VH + SZ_MTD_H;                 // 64*768 f16
static const size_t OFF_OH  = OFF_QH + (size_t)Bn * Dn * 2;      // 64*768 f16
static const size_t OFF_P   = OFF_OH + (size_t)Bn * Dn * 2;      // 64*768 f32
static const size_t OFF_WQT = OFF_P  + (size_t)Bn * Dn * 4;
static const size_t OFF_WKT = OFF_WQT + (size_t)Dn * Dn * 2;
static const size_t OFF_WVT = OFF_WKT + (size_t)Dn * Dn * 2;
static const size_t OFF_PJT = OFF_WVT + (size_t)Dn * Dn * 2;

// ---------------- WMMA fragment types ----------------
typedef __attribute__((ext_vector_type(16))) _Float16 v16h;
typedef __attribute__((ext_vector_type(8)))  float    v8f;
typedef __attribute__((ext_vector_type(4)))  unsigned int v4u;

union HFrag { v16h v; v4u q[2]; };

// =====================================================================
// Kernel: f32 [D,D] weight -> transposed f16 Wt[o][i]
// =====================================================================
__global__ void w_transpose_f16(const float* __restrict__ W, _Float16* __restrict__ Wt) {
    int idx = blockIdx.x * 256 + threadIdx.x;     // over D*D, idx = o*D + i
    int o = idx / Dn, i = idx - o * Dn;
    Wt[idx] = (_Float16)W[(size_t)i * Dn + o];
}

// =====================================================================
// Kernel: LayerNorm over concat rows -> n (f16), plus x_img passthrough
// one wave (32 lanes) per row, 8 rows per 256-thread block
// =====================================================================
__global__ void ln_prep(const float* __restrict__ xc, const float* __restrict__ xi,
                        const float* __restrict__ g, const float* __restrict__ bt,
                        _Float16* __restrict__ nh, float* __restrict__ out_img) {
    int wave = threadIdx.x >> 5, lane = threadIdx.x & 31;
    int r = blockIdx.x * 8 + wave;                // < 50240
    int b = r / Mn, t = r - b * Mn;
    const float* src = (t == 0) ? (xc + (size_t)b * Dn)
                                : (xi + ((size_t)b * Nn + (t - 1)) * Dn);
    float x[24]; float s = 0.f, s2 = 0.f;
    #pragma unroll
    for (int j = 0; j < 24; ++j) {
        float v = src[j * 32 + lane];
        x[j] = v; s += v; s2 += v * v;
    }
    #pragma unroll
    for (int o = 16; o; o >>= 1) { s += __shfl_xor(s, o, 32); s2 += __shfl_xor(s2, o, 32); }
    float mean = s * (1.f / 768.f);
    float var  = s2 * (1.f / 768.f) - mean * mean;
    float rinv = rsqrtf(var + 1e-6f);
    float* dst = (t == 0) ? nullptr : (out_img + ((size_t)b * Nn + (t - 1)) * Dn);
    _Float16* np = nh + (size_t)r * Dn;
    #pragma unroll
    for (int j = 0; j < 24; ++j) {
        int idx = j * 32 + lane;
        if (dst) dst[idx] = x[j];                           // passthrough copy
        np[idx] = (_Float16)((x[j] - mean) * rinv * g[idx] + bt[idx]);
    }
}

// =====================================================================
// Kernel: generic WMMA GEMM, C[M,768] = alpha*(A[M,768] @ Wt^T + bias)
//   A f16 (row stride lda elems), Wt f16 stored [o][i] (row-major, 768 K)
//   8 waves: wm in {0,1} x wn in {0..3}; wave tile 16M x 64N (4 acc)
//   WG tile 32M x 256N; K-step 64 (8 WMMA per barrier pair)
//   A tile staged memory->LDS with GLOBAL_LOAD_ASYNC_TO_LDS_B128 (ASYNCcnt)
//   grid = (768/256, Mrows/32)
// =====================================================================
__global__ void wmma_gemm768(const _Float16* __restrict__ A, long lda,
                             const _Float16* __restrict__ Wt,
                             const float* __restrict__ bias, float alpha,
                             _Float16* __restrict__ Ch, float* __restrict__ Cf,
                             int Mrows) {
    __shared__ __align__(16) _Float16 ta[32 * 64];        // A tile: 32 rows x 64 K (4 KB)
    int tid  = threadIdx.x;
    int wave = tid >> 5, lane = tid & 31;
    int wm = wave >> 2, wn = wave & 3;
    int l16 = lane & 15, lh = lane >> 4;
    int m0 = blockIdx.y * 32;
    int nbase = blockIdx.x * 256 + wn * 64;

    // each thread stages exactly one 16-byte chunk per K-step
    int rl = tid >> 3;          // 0..31  (row in tile)
    int ch = tid & 7;           // 0..7   (16B chunk within 64-half row)
    // LDS aperture: low 32 bits of the generic pointer == LDS byte offset
    unsigned loff = (unsigned)(size_t)&ta[rl * 64 + ch * 8];
    const _Float16* arow = A + (size_t)(m0 + rl) * lda + ch * 8;

    v8f acc[4];
    #pragma unroll
    for (int j = 0; j < 4; ++j) acc[j] = (v8f)0.f;

    for (int k0 = 0; k0 < Dn; k0 += 64) {
        __syncthreads();                                   // previous tile reads done
        {
            unsigned long long gaddr = (unsigned long long)(size_t)(arow + k0);
            // async DMA: memory -> LDS, tracked by ASYNCcnt (CDNA5)
            asm volatile("global_load_async_to_lds_b128 %0, %1, off"
                         :: "v"(loff), "v"(gaddr) : "memory");
            asm volatile("s_wait_asynccnt 0x0" ::: "memory");
        }
        __syncthreads();                                   // tile visible to all waves

        // A fragments per ISA 16-bit A 16x32 layout:
        //   lanes 0-15: VGPR0-3 K=0..7, VGPR4-7 K=16..23
        //   lanes16-31: VGPR0-3 K=8..15, VGPR4-7 K=24..31
        HFrag fa0, fa1;
        const _Float16* ap = &ta[(wm * 16 + l16) * 64 + lh * 8];
        fa0.q[0] = *(const v4u*)ap;
        fa0.q[1] = *(const v4u*)(ap + 16);
        fa1.q[0] = *(const v4u*)(ap + 32);
        fa1.q[1] = *(const v4u*)(ap + 48);

        #pragma unroll
        for (int j = 0; j < 4; ++j) {
            // B fragment (32x16): lanes 0-15 -> N=lane, K=0..7,16..23;
            //                     lanes16-31 -> N=lane-16, K=8..15,24..31
            HFrag fb0, fb1;
            const _Float16* wp = Wt + (size_t)(nbase + j * 16 + l16) * Dn + k0 + lh * 8;
            fb0.q[0] = *(const v4u*)wp;
            fb0.q[1] = *(const v4u*)(wp + 16);
            fb1.q[0] = *(const v4u*)(wp + 32);
            fb1.q[1] = *(const v4u*)(wp + 48);
            __builtin_prefetch((const void*)(wp + 64), 0, 3);   // next K chunk (L2-resident)
            acc[j] = __builtin_amdgcn_wmma_f32_16x16x32_f16(
                         false, fa0.v, false, fb0.v, (short)0, acc[j], false, false);
            acc[j] = __builtin_amdgcn_wmma_f32_16x16x32_f16(
                         false, fa1.v, false, fb1.v, (short)0, acc[j], false, false);
        }
    }

    // epilogue: C/D 16x16 f32 layout — lane 0-15: N=lane, VGPR r: M=r;
    //           lane16-31: N=lane-16, VGPR r: M=r+8
    #pragma unroll
    for (int j = 0; j < 4; ++j) {
        int n = nbase + j * 16 + l16;
        float bv = bias[n];
        #pragma unroll
        for (int r = 0; r < 8; ++r) {
            int m = m0 + wm * 16 + lh * 8 + r;
            if (m < Mrows) {
                float val = alpha * (acc[j][r] + bv);
                if (Ch) Ch[(size_t)m * Dn + n] = (_Float16)val;
                else    Cf[(size_t)m * Dn + n] = val;
            }
        }
    }
}

// =====================================================================
// Kernel: class attention, one 256-thread WG per batch.
// score matrix [32 heads][785] lives in dynamic LDS (~107 KB; WGP has 320 KB)
// =====================================================================
#define SROW 800
__global__ void attn_kernel(const _Float16* __restrict__ qh,
                            const _Float16* __restrict__ kh,
                            const _Float16* __restrict__ vh,
                            const float* __restrict__ w1W, const float* __restrict__ w1b,
                            const float* __restrict__ w2W, const float* __restrict__ w2b,
                            _Float16* __restrict__ oh) {
    extern __shared__ float sm[];
    float* sS  = sm;                    // 32*800
    float* sQ  = sm + 32 * SROW;        // 768
    float* sW  = sQ + Dn;               // 1024
    float* sWb = sW + 1024;             // 32

    int b = blockIdx.x, tid = threadIdx.x;
    int wave = tid >> 5, lane = tid & 31;

    for (int d = tid; d < Dn; d += 256) sQ[d] = (float)qh[(size_t)b * Dn + d];
    for (int i = tid; i < 1024; i += 256) sW[i] = w1W[i];
    if (tid < 32) sWb[tid] = w1b[tid];
    __syncthreads();

    // scores: s[h][m] = q_h . k_h  (q already scaled). lane = head, wave = m strip.
    for (int m = wave; m < Mn; m += 8) {
        const _Float16* krow = kh + ((size_t)b * Mn + m) * Dn + lane * HDn;
        const float* qv = sQ + lane * HDn;
        float acc = 0.f;
        #pragma unroll
        for (int d = 0; d < HDn; ++d) acc += qv[d] * (float)krow[d];
        sS[lane * SROW + m] = acc;
    }
    __syncthreads();

    // pre-softmax talking heads: s'[g] = s[g] + sum_h s[h]*w1[h][g] + w1_b[g]
    for (int m = tid; m < Mn; m += 256) {
        float col[32];
        #pragma unroll
        for (int h = 0; h < 32; ++h) col[h] = sS[h * SROW + m];
        for (int g2 = 0; g2 < 32; ++g2) {
            float o = col[g2] + sWb[g2];
            #pragma unroll
            for (int h = 0; h < 32; ++h) o += col[h] * sW[h * 32 + g2];
            sS[g2 * SROW + m] = o;
        }
    }
    __syncthreads();

    // stage w2 (sW reads from pre-mix are done); reads happen after next barrier
    for (int i = tid; i < 1024; i += 256) sW[i] = w2W[i];
    if (tid < 32) sWb[tid] = w2b[tid];

    // softmax over m, one wave per head row
    for (int h = wave; h < 32; h += 8) {
        float mx = -3.4e38f;
        for (int m = lane; m < Mn; m += 32) mx = fmaxf(mx, sS[h * SROW + m]);
        #pragma unroll
        for (int o = 16; o; o >>= 1) mx = fmaxf(mx, __shfl_xor(mx, o, 32));
        float sum = 0.f;
        for (int m = lane; m < Mn; m += 32) {
            float e = __expf(sS[h * SROW + m] - mx);
            sS[h * SROW + m] = e; sum += e;
        }
        #pragma unroll
        for (int o = 16; o; o >>= 1) sum += __shfl_xor(sum, o, 32);
        float inv = 1.f / sum;
        for (int m = lane; m < Mn; m += 32) sS[h * SROW + m] *= inv;
    }
    __syncthreads();

    // post-softmax talking heads (w2)
    for (int m = tid; m < Mn; m += 256) {
        float col[32];
        #pragma unroll
        for (int h = 0; h < 32; ++h) col[h] = sS[h * SROW + m];
        for (int g2 = 0; g2 < 32; ++g2) {
            float o = col[g2] + sWb[g2];
            #pragma unroll
            for (int h = 0; h < 32; ++h) o += col[h] * sW[h * 32 + g2];
            sS[g2 * SROW + m] = o;
        }
    }
    __syncthreads();

    // o[h][d] = sum_m attn[h][m] * v[m][h*24+d]   (threads cover 768 d's, 3 each)
    float acc[3]; int hh[3], dd[3];
    #pragma unroll
    for (int r = 0; r < 3; ++r) { dd[r] = tid + 256 * r; hh[r] = dd[r] / HDn; acc[r] = 0.f; }
    for (int m = 0; m < Mn; ++m) {
        const _Float16* vrow = vh + ((size_t)b * Mn + m) * Dn;
        #pragma unroll
        for (int r = 0; r < 3; ++r) acc[r] += sS[hh[r] * SROW + m] * (float)vrow[dd[r]];
    }
    #pragma unroll
    for (int r = 0; r < 3; ++r) oh[(size_t)b * Dn + dd[r]] = (_Float16)acc[r];
}

// =====================================================================
// Kernel: cls tail — residual+proj result, LN2, grouped fc1, exact GELU,
// channel shuffle, grouped fc2, residual. One 256-thread WG per batch.
// =====================================================================
__global__ void mlp_kernel(const float* __restrict__ xc, const float* __restrict__ P,
                           const float* __restrict__ g2, const float* __restrict__ b2,
                           const float* __restrict__ fc1W, const float* __restrict__ fc1b,
                           const float* __restrict__ fc2W, const float* __restrict__ fc2b,
                           float* __restrict__ out_cls) {
    __shared__ float cbuf[Dn], ybuf[Dn], hbuf[HIDn], red[256];
    int b = blockIdx.x, tid = threadIdx.x;

    float local = 0.f;
    #pragma unroll
    for (int r = 0; r < 3; ++r) {
        int d = tid + 256 * r;
        float v = xc[(size_t)b * Dn + d] + P[(size_t)b * Dn + d];  // P = o@proj + proj_b
        cbuf[d] = v; local += v;
    }
    red[tid] = local; __syncthreads();
    for (int s = 128; s; s >>= 1) { if (tid < s) red[tid] += red[tid + s]; __syncthreads(); }
    float mean = red[0] * (1.f / 768.f);
    __syncthreads();
    local = 0.f;
    #pragma unroll
    for (int r = 0; r < 3; ++r) { int d = tid + 256 * r; float v = cbuf[d] - mean; local += v * v; }
    red[tid] = local; __syncthreads();
    for (int s = 128; s; s >>= 1) { if (tid < s) red[tid] += red[tid + s]; __syncthreads(); }
    float rinv = rsqrtf(red[0] * (1.f / 768.f) + 1e-6f);
    #pragma unroll
    for (int r = 0; r < 3; ++r) {
        int d = tid + 256 * r;
        ybuf[d] = (cbuf[d] - mean) * rinv * g2[d] + b2[d];
    }
    __syncthreads();

    // fc1 (grouped) + exact GELU
    for (int j = tid; j < HIDn; j += 256) {
        int gg = j / (HIDn / Gn), oo = j - gg * (HIDn / Gn);
        const float* wrow = fc1W + ((size_t)gg * (HIDn / Gn) + oo) * (Dn / Gn);
        const float* yv = ybuf + gg * (Dn / Gn);
        float a = fc1b[gg * (HIDn / Gn) + oo];
        for (int c = 0; c < Dn / Gn; ++c) a += yv[c] * wrow[c];
        hbuf[j] = 0.5f * a * (1.f + erff(a * 0.70710678118654752f));
    }
    __syncthreads();

    // channel shuffle + fc2 (grouped) + residual
    for (int j = tid; j < Dn; j += 256) {
        int gg = j / (Dn / Gn), oo = j - gg * (Dn / Gn);
        const float* wrow = fc2W + ((size_t)gg * (Dn / Gn) + oo) * (HIDn / Gn);
        float a = fc2b[gg * (Dn / Gn) + oo];
        for (int c = 0; c < HIDn / Gn; ++c) a += hbuf[c * Gn + gg] * wrow[c];  // shuffled read
        out_cls[(size_t)b * Dn + j] = cbuf[j] + a;
    }
}

// =====================================================================
extern "C" void kernel_launch(void* const* d_in, const int* in_sizes, int n_in,
                              void* d_out, int out_size, void* d_ws, size_t ws_size,
                              hipStream_t stream) {
    const float* x_cls = (const float*)d_in[0];
    const float* x_img = (const float*)d_in[1];
    const float* ln1_g = (const float*)d_in[2];
    const float* ln1_b = (const float*)d_in[3];
    const float* Wq    = (const float*)d_in[4];
    const float* bq    = (const float*)d_in[5];
    const float* Wk    = (const float*)d_in[6];
    const float* bk    = (const float*)d_in[7];
    const float* Wv    = (const float*)d_in[8];
    const float* bv    = (const float*)d_in[9];
    const float* w1W   = (const float*)d_in[10];
    const float* w1b   = (const float*)d_in[11];
    const float* w2W   = (const float*)d_in[12];
    const float* w2b   = (const float*)d_in[13];
    const float* projW = (const float*)d_in[14];
    const float* projb = (const float*)d_in[15];
    const float* ln2_g = (const float*)d_in[16];
    const float* ln2_b = (const float*)d_in[17];
    const float* fc1W  = (const float*)d_in[18];
    const float* fc1b  = (const float*)d_in[19];
    const float* fc2W  = (const float*)d_in[20];
    const float* fc2b  = (const float*)d_in[21];

    char* ws = (char*)d_ws;
    _Float16* n_h  = (_Float16*)(ws + OFF_NH);
    _Float16* k_h  = (_Float16*)(ws + OFF_KH);
    _Float16* v_h  = (_Float16*)(ws + OFF_VH);
    _Float16* q_h  = (_Float16*)(ws + OFF_QH);
    _Float16* o_h  = (_Float16*)(ws + OFF_OH);
    float*    Pbuf = (float*)   (ws + OFF_P);
    _Float16* WqT  = (_Float16*)(ws + OFF_WQT);
    _Float16* WkT  = (_Float16*)(ws + OFF_WKT);
    _Float16* WvT  = (_Float16*)(ws + OFF_WVT);
    _Float16* PjT  = (_Float16*)(ws + OFF_PJT);

    float* out_cls = (float*)d_out;
    float* out_img = out_cls + (size_t)Bn * Dn;

    // weight conversions (independent)
    int wblk = (Dn * Dn) / 256;   // 2304
    w_transpose_f16<<<wblk, 256, 0, stream>>>(Wq,    WqT);
    w_transpose_f16<<<wblk, 256, 0, stream>>>(Wk,    WkT);
    w_transpose_f16<<<wblk, 256, 0, stream>>>(Wv,    WvT);
    w_transpose_f16<<<wblk, 256, 0, stream>>>(projW, PjT);

    // LN + f16 pack + passthrough copy (50240 rows, 8 per block)
    ln_prep<<<MTn / 8, 256, 0, stream>>>(x_cls, x_img, ln1_g, ln1_b, n_h, out_img);

    // K / V : 50240 x 768 x 768 f16 WMMA GEMMs
    dim3 gKV(Dn / 256, MTn / 32);          // (3, 1570)
    wmma_gemm768<<<gKV, 256, 0, stream>>>(n_h, (long)Dn, WkT, bk, 1.0f, k_h, nullptr, MTn);
    wmma_gemm768<<<gKV, 256, 0, stream>>>(n_h, (long)Dn, WvT, bv, 1.0f, v_h, nullptr, MTn);

    // Q : cls rows only (row b of A = n_h + b*785*768), alpha = HD^-0.5
    dim3 gQ(Dn / 256, Bn / 32);            // (3, 2)
    wmma_gemm768<<<gQ, 256, 0, stream>>>(n_h, (long)Mn * Dn, WqT, bq, SCALEq, q_h, nullptr, Bn);

    // attention (dynamic LDS ~107 KB; CDNA5 WGP has 320 KB)
    size_t smem = (size_t)(32 * SROW + Dn + 1024 + 32) * sizeof(float);
    hipFuncSetAttribute((const void*)attn_kernel,
                        hipFuncAttributeMaxDynamicSharedMemorySize, (int)smem);
    attn_kernel<<<Bn, 256, smem, stream>>>(q_h, k_h, v_h, w1W, w1b, w2W, w2b, o_h);

    // proj : P = o @ projW + proj_b  (f32 output)
    wmma_gemm768<<<gQ, 256, 0, stream>>>(o_h, (long)Dn, PjT, projb, 1.0f, nullptr, Pbuf, Bn);

    // cls tail MLP + residuals -> out_cls
    mlp_kernel<<<Bn, 256, 0, stream>>>(x_cls, Pbuf, ln2_g, ln2_b,
                                       fc1W, fc1b, fc2W, fc2b, out_cls);
}
